// MambaBlock_9405978378870
// MI455X (gfx1250) — matrix operand order repaired and checked by reading, not compiled, not run on hardware
//
#include <hip/hip_runtime.h>
#include <hip/hip_bf16.h>
#include <math.h>

// ---------------- problem constants (from reference) ----------------
#define D_MODEL   2048
#define D_INNER   4096
#define D_STATE   16
#define D_CONV    4
#define DT_RANK   128
#define BATCH     2
#define SEQ       512
#define NTOK      (BATCH * SEQ)          // 1024 token rows
#define RESET_DT  10000.0f

typedef __attribute__((ext_vector_type(16))) __bf16 v16bf;
typedef __attribute__((ext_vector_type(8)))  float  v8f;
typedef __attribute__((ext_vector_type(2)))  float  v2f;

// =====================================================================
// bf16 WMMA GEMM:  C[M,N] = A[M,K] * W[N,K]^T   (all fp32 in memory,
// fp32->bf16 conversion fused into fragment load, fp32 accumulate).
// One wave computes a 16(M) x 64(N) tile (A fragment reused 4x).
// A-frag (16x32 bf16): lane half h=lane>>4 holds K[16h..16h+16)
// of row (lane&15)  -> one contiguous 16-float run.
// B-frag (32x16 bf16): lane holds K[16h..16h+16) of W row n0+(lane&15)
// -> one contiguous 16-float run.
// =====================================================================
__global__ __launch_bounds__(256)
void gemm_bf16_wmma(const float* __restrict__ A, int lda,
                    const float* __restrict__ W, int ldw,
                    float* __restrict__ C, int ldc,
                    int M, int N, int K)
{
    const int lane  = threadIdx.x & 31;
    const int wave  = threadIdx.x >> 5;
    const int tilesN = N >> 6;                       // N/64
    const int tile   = blockIdx.x * 8 + wave;        // 8 waves / block
    const int tm = tile / tilesN;
    const int tn = tile - tm * tilesN;
    const int m0 = tm << 4;
    const int n0 = tn << 6;
    const int half = lane >> 4;
    const int l15  = lane & 15;

    const float* arow = A + (size_t)(m0 + l15) * lda + half * 16;
    const float* wrow0 = W + (size_t)(n0 +  0 + l15) * ldw + half * 16;
    const float* wrow1 = W + (size_t)(n0 + 16 + l15) * ldw + half * 16;
    const float* wrow2 = W + (size_t)(n0 + 32 + l15) * ldw + half * 16;
    const float* wrow3 = W + (size_t)(n0 + 48 + l15) * ldw + half * 16;

    v8f acc[4] = {};

    for (int k0 = 0; k0 < K; k0 += 32) {
        v16bf a;
#pragma unroll
        for (int i = 0; i < 8; ++i)  a[i]     = (__bf16)arow[k0 + i];
#pragma unroll
        for (int i = 0; i < 8; ++i)  a[8 + i] = (__bf16)arow[k0 + 8 + i];

        v16bf b0, b1, b2, b3;
#pragma unroll
        for (int i = 0; i < 16; ++i) b0[i] = (__bf16)wrow0[k0 + i];
#pragma unroll
        for (int i = 0; i < 16; ++i) b1[i] = (__bf16)wrow1[k0 + i];
#pragma unroll
        for (int i = 0; i < 16; ++i) b2[i] = (__bf16)wrow2[k0 + i];
#pragma unroll
        for (int i = 0; i < 16; ++i) b3[i] = (__bf16)wrow3[k0 + i];

        acc[0] = __builtin_amdgcn_wmma_f32_16x16x32_bf16(false, a, false, b0,
                                                         (short)0, acc[0], false, false);
        acc[1] = __builtin_amdgcn_wmma_f32_16x16x32_bf16(false, a, false, b1,
                                                         (short)0, acc[1], false, false);
        acc[2] = __builtin_amdgcn_wmma_f32_16x16x32_bf16(false, a, false, b2,
                                                         (short)0, acc[2], false, false);
        acc[3] = __builtin_amdgcn_wmma_f32_16x16x32_bf16(false, a, false, b3,
                                                         (short)0, acc[3], false, false);
    }

    // C/D layout: VGPR p -> row m0 + p + 8*half, col n0 + 16*j + (lane&15)
#pragma unroll
    for (int j = 0; j < 4; ++j) {
        float* crow = C + (size_t)(m0 + half * 8) * ldc + n0 + 16 * j + l15;
#pragma unroll
        for (int p = 0; p < 8; ++p)
            crow[(size_t)p * ldc] = acc[j][p];
    }
}

// =====================================================================
// fp32 WMMA GEMM (16x16x4): exact-precision path for small projections.
// One wave computes a 16x16 tile.  A-frag: lane holds A[row][k0+2h..+2)
// B-frag: lane holds W[n][k0+2h..+2)  (both contiguous float2).
// =====================================================================
__global__ __launch_bounds__(256)
void gemm_f32_wmma(const float* __restrict__ A, int lda,
                   const float* __restrict__ W, int ldw,
                   float* __restrict__ C, int ldc,
                   int M, int N, int K)
{
    const int lane = threadIdx.x & 31;
    const int wave = threadIdx.x >> 5;
    const int tilesN = N >> 4;
    const int tile = blockIdx.x * 8 + wave;
    const int tm = tile / tilesN;
    const int tn = tile - tm * tilesN;
    const int m0 = tm << 4;
    const int n0 = tn << 4;
    const int half = lane >> 4;
    const int l15  = lane & 15;

    const float* ap = A + (size_t)(m0 + l15) * lda + half * 2;
    const float* wp = W + (size_t)(n0 + l15) * ldw + half * 2;

    v8f acc = {};
    for (int k0 = 0; k0 < K; k0 += 4) {
        v2f a; a[0] = ap[k0]; a[1] = ap[k0 + 1];
        v2f b; b[0] = wp[k0]; b[1] = wp[k0 + 1];
        acc = __builtin_amdgcn_wmma_f32_16x16x4_f32(false, a, false, b,
                                                    (short)0, acc, false, false);
    }

    float* crow = C + (size_t)(m0 + half * 8) * ldc + n0 + l15;
#pragma unroll
    for (int p = 0; p < 8; ++p)
        crow[(size_t)p * ldc] = acc[p];
}

// =====================================================================
// dt projection (fp32 WMMA) with fused epilogue:
//   dt = max(softplus(acc + b_dt), 1e-5) ; reset rows -> RESET_DT
// A = x_dbl[:, 0:128] (lda=160), W = W_dt (4096x128), out = dt (1024x4096)
// =====================================================================
__global__ __launch_bounds__(256)
void gemm_dt_wmma(const float* __restrict__ A, int lda,
                  const float* __restrict__ W, int ldw,
                  const float* __restrict__ bias,
                  const unsigned char* __restrict__ reset,
                  float* __restrict__ C, int ldc,
                  int M, int N, int K)
{
    const int lane = threadIdx.x & 31;
    const int wave = threadIdx.x >> 5;
    const int tilesN = N >> 4;
    const int tile = blockIdx.x * 8 + wave;
    const int tm = tile / tilesN;
    const int tn = tile - tm * tilesN;
    const int m0 = tm << 4;
    const int n0 = tn << 4;
    const int half = lane >> 4;
    const int l15  = lane & 15;

    const float* ap = A + (size_t)(m0 + l15) * lda + half * 2;
    const float* wp = W + (size_t)(n0 + l15) * ldw + half * 2;

    v8f acc = {};
    for (int k0 = 0; k0 < K; k0 += 4) {
        v2f a; a[0] = ap[k0]; a[1] = ap[k0 + 1];
        v2f b; b[0] = wp[k0]; b[1] = wp[k0 + 1];
        acc = __builtin_amdgcn_wmma_f32_16x16x4_f32(false, a, false, b,
                                                    (short)0, acc, false, false);
    }

    const int   col = n0 + l15;
    const float bn  = bias[col];
#pragma unroll
    for (int p = 0; p < 8; ++p) {
        int m = m0 + half * 8 + p;                // token row
        float v  = acc[p] + bn;
        float sp = (v > 20.0f) ? v : log1pf(expf(v));
        float dt = fmaxf(sp, 1e-5f);
        if (reset[m]) dt = RESET_DT;
        C[(size_t)m * ldc + col] = dt;
    }
}

// =====================================================================
// Depthwise causal conv1d (win=4) + bias + SiLU + reset mask.
// xz: (1024, 8192) from in_proj; x-branch = cols [0,4096).
// =====================================================================
__global__ __launch_bounds__(256)
void conv_silu(const float* __restrict__ xz,
               const float* __restrict__ conv_w,
               const float* __restrict__ conv_b,
               const unsigned char* __restrict__ reset,
               float* __restrict__ xconv)
{
    const int idx = blockIdx.x * 256 + threadIdx.x;   // over 1024*4096
    const int d = idx & (D_INNER - 1);
    const int m = idx >> 12;                          // token row 0..1023
    const int b = m >> 9;
    const int t = m & (SEQ - 1);

    float acc = conv_b[d];
#pragma unroll
    for (int k = 0; k < D_CONV; ++k) {
        int tt = t - (D_CONV - 1) + k;
        if (tt >= 0)
            acc += conv_w[d * D_CONV + k] *
                   xz[((size_t)(b * SEQ + tt)) * (2 * D_INNER) + d];
    }
    float s = acc / (1.0f + __expf(-acc)) ;           // silu = a*sigmoid(a)
    if (reset[m]) s = 0.0f;                           // keep-mask
    xconv[(size_t)m * D_INNER + d] = s;
}

// =====================================================================
// Fused selective-scan. One lane owns one (batch, d_inner) channel,
// h[16] lives in VGPRs.  dA = exp(dt*A) and dBx = dt*B*x are computed
// on the fly (never materialized: saves ~1 GB of HBM traffic vs the
// reference dataflow).  Epilogue fuses +x*D and the SiLU(z) gate.
// =====================================================================
__global__ __launch_bounds__(256)
void ssm_scan(const float* __restrict__ xconv,
              const float* __restrict__ xz,       // z = cols [4096,8192)
              const float* __restrict__ xdbl,     // (1024,160): [dt_low|B|C]
              const float* __restrict__ dtb,      // (1024,4096)
              const float* __restrict__ A_log,    // (4096,16)
              const float* __restrict__ Dvec,     // (4096)
              const unsigned char* __restrict__ reset,
              float* __restrict__ yg)             // (1024,4096)
{
    const int ch = blockIdx.x * 256 + threadIdx.x;   // 0..8191
    const int b  = ch >> 12;
    const int d  = ch & (D_INNER - 1);

    float An[D_STATE];
#pragma unroll
    for (int n = 0; n < D_STATE; ++n)
        An[n] = -__expf(A_log[d * D_STATE + n]);
    const float Dd = Dvec[d];

    float h[D_STATE];
#pragma unroll
    for (int n = 0; n < D_STATE; ++n) h[n] = 0.0f;

    for (int t = 0; t < SEQ; ++t) {
        const size_t m  = (size_t)b * SEQ + t;
        const float dt  = dtb[m * D_INNER + d];
        const float x   = xconv[m * D_INNER + d];
        const float* BC = xdbl + m * (DT_RANK + 2 * D_STATE) + DT_RANK; // wave-uniform
        const float dtx = dt * x;

        float y = 0.0f;
#pragma unroll
        for (int n = 0; n < D_STATE; ++n) {
            float dA = __expf(dt * An[n]);
            h[n] = dA * h[n] + dtx * BC[n];          // dA*h + dt*B*x
            y   += h[n] * BC[D_STATE + n];           // * C
        }

        float z = xz[m * (2 * D_INNER) + D_INNER + d];
        if (reset[m]) z = 0.0f;
        float gate = z / (1.0f + __expf(-z));
        yg[m * D_INNER + d] = (y + x * Dd) * gate;
    }
}

// =====================================================================
// Host orchestration
// =====================================================================
extern "C" void kernel_launch(void* const* d_in, const int* in_sizes, int n_in,
                              void* d_out, int out_size, void* d_ws, size_t ws_size,
                              hipStream_t stream)
{
    const float*         x      = (const float*)d_in[0];
    const unsigned char* reset  = (const unsigned char*)d_in[1]; // all-false bools
    const float*         W_in   = (const float*)d_in[2];
    const float*         conv_w = (const float*)d_in[3];
    const float*         conv_b = (const float*)d_in[4];
    const float*         W_x    = (const float*)d_in[5];
    const float*         W_dt   = (const float*)d_in[6];
    const float*         b_dt   = (const float*)d_in[7];
    const float*         A_log  = (const float*)d_in[8];
    const float*         Dvec   = (const float*)d_in[9];
    const float*         W_out  = (const float*)d_in[10];
    float*               out    = (float*)d_out;

    // workspace partition (floats)
    float* ws    = (float*)d_ws;
    float* xz    = ws;                                   // 1024 x 8192  (32 MB)
    float* xconv = xz    + (size_t)NTOK * 2 * D_INNER;   // 1024 x 4096  (16 MB)
    float* xdbl  = xconv + (size_t)NTOK * D_INNER;       // 1024 x 160
    float* dtb   = xdbl  + (size_t)NTOK * (DT_RANK + 2 * D_STATE); // 1024 x 4096
    float* yg    = dtb   + (size_t)NTOK * D_INNER;       // 1024 x 4096

    // 1) in_proj: xz = x @ W_in^T          (bf16 WMMA, 34.4 GFLOP)
    //    tiles = (1024/16)*(8192/64) = 8192 -> 1024 blocks x 8 waves
    gemm_bf16_wmma<<<1024, 256, 0, stream>>>(x, D_MODEL, W_in, D_MODEL,
                                             xz, 2 * D_INNER,
                                             NTOK, 2 * D_INNER, D_MODEL);

    // 2) causal depthwise conv + SiLU + mask
    conv_silu<<<(NTOK * D_INNER) / 256, 256, 0, stream>>>(xz, conv_w, conv_b,
                                                          reset, xconv);

    // 3) x_dbl = xconv @ W_x^T             (fp32 WMMA, N=160)
    //    tiles = 64*10 = 640 -> 80 blocks
    gemm_f32_wmma<<<80, 256, 0, stream>>>(xconv, D_INNER, W_x, D_INNER,
                                          xdbl, DT_RANK + 2 * D_STATE,
                                          NTOK, DT_RANK + 2 * D_STATE, D_INNER);

    // 4) dt = softplus(dt_low @ W_dt^T + b_dt)   (fp32 WMMA + epilogue)
    //    tiles = 64*256 = 16384 -> 2048 blocks
    gemm_dt_wmma<<<2048, 256, 0, stream>>>(xdbl, DT_RANK + 2 * D_STATE,
                                           W_dt, DT_RANK, b_dt, reset,
                                           dtb, D_INNER,
                                           NTOK, D_INNER, DT_RANK);

    // 5) fused selective scan + D-skip + SiLU(z) gate
    ssm_scan<<<(BATCH * D_INNER) / 256, 256, 0, stream>>>(xconv, xz, xdbl, dtb,
                                                          A_log, Dvec, reset, yg);

    // 6) out_proj: out = yg @ W_out^T      (bf16 WMMA, 17.2 GFLOP)
    //    tiles = 64*32 = 2048 -> 256 blocks
    gemm_bf16_wmma<<<256, 256, 0, stream>>>(yg, D_INNER, W_out, D_INNER,
                                            out, D_MODEL,
                                            NTOK, D_MODEL, D_INNER);
}
